// TransformerModuleNet_14894946582989
// MI455X (gfx1250) — compile-verified, matrix operand order checked by reading, not run on hardware
//
#include <hip/hip_runtime.h>
#include <hip/hip_bf16.h>

// ---------------------------------------------------------------------------
// TransformerModuleNet for MI455X (gfx1250, wave32, WMMA).
// - All weights pre-packed to bf16 WMMA B-fragment layout (halves the
//   weight bandwidth that dominates this workload; fragment loads are
//   contiguous 128-bit and fully coalesced).
// - Every GEMM runs through v_wmma_f32_16x16x32_bf16, fp32 accumulate.
// - Each wave computes a 16x32 output strip (2 N-tiles share one A fragment)
//   with a branch-free inner loop: buffers are padded to 16-row tiles so no
//   load/store masking is needed (pad output rows are never consumed).
// - The data-dependent program (args) is resolved on-device (grid.z = batch)
//   so the whole launch is graph-capturable with fixed grids.
// ---------------------------------------------------------------------------

#define HID    768
#define HEADS  12
#define HD     64
#define FF     3072
#define VFEAT  2048
#define LTOK   36
#define BS     8
#define NMOD   26
#define NSTEP  25

typedef __attribute__((ext_vector_type(16))) __bf16 v16bf;
typedef __attribute__((ext_vector_type(8)))  float  v8f;

// ---------------- workspace layout (element offsets) -----------------------
// ushort (bf16 storage) region at byte 0, float region after FBYTE_BASE.
static constexpr size_t WSZ_QKVAO = 768ull * 768;          // 589824
static constexpr size_t WSZ_FF    = 768ull * 3072;         // 2359296
static constexpr size_t WQ  = 0;
static constexpr size_t WK  = 1 * WSZ_QKVAO;
static constexpr size_t WV  = 2 * WSZ_QKVAO;
static constexpr size_t WAO = 3 * WSZ_QKVAO;
static constexpr size_t WI  = 4 * WSZ_QKVAO;
static constexpr size_t WO  = 4 * WSZ_QKVAO + WSZ_FF;
static constexpr size_t WSLAB = 4 * WSZ_QKVAO + 2 * WSZ_FF; // 7077888 per module

static constexpr size_t OFF_WPACK    = 0;
static constexpr size_t OFF_IMGWP    = OFF_WPACK + (size_t)NMOD * WSLAB;
static constexpr size_t OFF_CLS1P    = OFF_IMGWP + (size_t)VFEAT * HID;
static constexpr size_t OFF_CLS2P    = OFF_CLS1P + (size_t)HID * (2 * HID);
static constexpr size_t OFF_FEATBF   = OFF_CLS2P + (size_t)(2 * HID) * 32;
static constexpr size_t OFF_XBF      = OFF_FEATBF + (size_t)BS * LTOK * VFEAT;
static constexpr size_t OFF_CTXBF    = OFF_XBF + (size_t)BS * 80 * HID;
static constexpr size_t OFF_H1BF     = OFF_CTXBF + (size_t)BS * 48 * HID;
static constexpr size_t OFF_FFBF     = OFF_H1BF + (size_t)BS * 48 * HID;
static constexpr size_t OFF_CLSINBF  = OFF_FFBF + (size_t)BS * 48 * FF;
static constexpr size_t OFF_CLSMIDBF = OFF_CLSINBF + 16ull * HID;
static constexpr size_t USHORT_TOTAL = OFF_CLSMIDBF + 16ull * (2 * HID);
static constexpr size_t FBYTE_BASE   = ((USHORT_TOTAL * 2 + 255) / 256) * 256;

// float region
static constexpr size_t PSLABF   = 9984; // biases + LN params per module
static constexpr size_t PB_QB = 0, PB_KB = 768, PB_VB = 1536, PB_AOB = 2304;
static constexpr size_t PB_IB = 3072, PB_OB = 6144;
static constexpr size_t PB_L1G = 6912, PB_L1B = 7680, PB_L2G = 8448, PB_L2B = 9216;

static constexpr size_t OFF_PBLK = 0;
static constexpr size_t OFF_HS   = OFF_PBLK + (size_t)NMOD * PSLABF;      // BS x 26 x 36 x 768
static constexpr size_t OFF_XF   = OFF_HS + (size_t)BS * 26 * LTOK * HID;
static constexpr size_t OFF_QKV  = OFF_XF + (size_t)BS * 80 * HID;
static constexpr size_t OFF_AOUT = OFF_QKV + (size_t)BS * 80 * (3 * HID);
static constexpr size_t OFF_H1F  = OFF_AOUT + (size_t)BS * 48 * HID;
static constexpr size_t OFF_OOUT = OFF_H1F + (size_t)BS * 48 * HID;
static constexpr size_t OFF_EMBT = OFF_OOUT + (size_t)BS * 48 * HID;      // 288 x 768
static constexpr size_t OFF_PRED = OFF_EMBT + (size_t)BS * LTOK * HID;    // 16 x 32

// ---------------------------- device helpers -------------------------------
__device__ __forceinline__ unsigned short* Ureg(void* ws) {
  return (unsigned short*)ws;
}
__device__ __forceinline__ float* Freg(void* ws) {
  return (float*)((char*)ws + FBYTE_BASE);
}

__device__ __forceinline__ unsigned short f2bf(float f) {
  unsigned u = __float_as_uint(f);
  unsigned r = u + 0x7FFFu + ((u >> 16) & 1u);   // round-to-nearest-even
  return (unsigned short)(r >> 16);
}

__device__ __forceinline__ float gelu_exact(float x) {
  return 0.5f * x * (1.0f + erff(x * 0.70710678118654752440f));
}

union Frag { uint4 u[2]; v16bf v; };

// One 16x32 D strip per wave (two N-tiles sharing one A fragment), K-loop in
// steps of 32 using v_wmma_f32_16x16x32_bf16.  Branch-free inner loop: all
// buffers are padded to whole 16-row tiles, so no load/store masking.
// A is row-major bf16 [tiles*16 x K] (lda == K).
// Bp is pre-packed fragment layout: ((ntile*(K/32)+ks)*32 + lane)*16 + e.
__device__ __forceinline__ void gemm_wave_body(
    const unsigned short* __restrict__ A, const unsigned short* __restrict__ Bp,
    const float* __restrict__ bias, float* __restrict__ Cf,
    unsigned short* __restrict__ Cbf, int K, int N, int ldc,
    int coloff, int act) {
  int wave = threadIdx.x >> 5;
  int lane = threadIdx.x & 31;
  int pair = blockIdx.x * (blockDim.x >> 5) + wave;  // strip of 2 N-tiles
  int nt0  = pair * 2;
  if (nt0 * 16 >= N) return;                         // wave-uniform exit
  int mtile = blockIdx.y;
  int half  = lane >> 4;
  int nl    = lane & 15;

  const unsigned short* arow = A + (size_t)(mtile * 16 + nl) * K;
  int ksteps = K >> 5;
  const unsigned short* bptr0 =
      Bp + (size_t)nt0 * ksteps * 512 + (size_t)lane * 16;
  const unsigned short* bptr1 = bptr0 + (size_t)ksteps * 512;

  v8f acc0 = {0.f, 0.f, 0.f, 0.f, 0.f, 0.f, 0.f, 0.f};
  v8f acc1 = {0.f, 0.f, 0.f, 0.f, 0.f, 0.f, 0.f, 0.f};
  for (int ks = 0; ks < ksteps; ++ks) {
    Frag fa, fb0, fb1;
    fa.u[0]  = *(const uint4*)(arow + ks * 32 + half * 8);
    fa.u[1]  = *(const uint4*)(arow + ks * 32 + 16 + half * 8);
    fb0.u[0] = *(const uint4*)(bptr0);
    fb0.u[1] = *(const uint4*)(bptr0 + 8);
    fb1.u[0] = *(const uint4*)(bptr1);
    fb1.u[1] = *(const uint4*)(bptr1 + 8);
    bptr0 += 512;
    bptr1 += 512;
    acc0 = __builtin_amdgcn_wmma_f32_16x16x32_bf16(
        false, fa.v, false, fb0.v, (short)0, acc0, false, false);
    acc1 = __builtin_amdgcn_wmma_f32_16x16x32_bf16(
        false, fa.v, false, fb1.v, (short)0, acc1, false, false);
  }

  float bv0 = bias ? bias[nt0 * 16 + nl] : 0.f;
  float bv1 = bias ? bias[nt0 * 16 + 16 + nl] : 0.f;
  int col0 = coloff + nt0 * 16 + nl;
#pragma unroll
  for (int i = 0; i < 8; ++i) {
    int r = mtile * 16 + half * 8 + i;
    float v0 = acc0[i] + bv0;
    float v1 = acc1[i] + bv1;
    if (act == 1)      { v0 = gelu_exact(v0); v1 = gelu_exact(v1); }
    else if (act == 2) { v0 = fmaxf(v0, 0.f); v1 = fmaxf(v1, 0.f); }
    if (Cbf) {
      Cbf[(size_t)r * ldc + col0]      = f2bf(v0);
      Cbf[(size_t)r * ldc + col0 + 16] = f2bf(v1);
    } else {
      Cf[(size_t)r * ldc + col0]      = v0;
      Cf[(size_t)r * ldc + col0 + 16] = v1;
    }
  }
}

// ------------------------------ pack kernels -------------------------------
__device__ __forceinline__ void pack_one(const float* src, unsigned short* dst,
                                         size_t local, int K, int N) {
  int e    = (int)(local & 15);
  int lane = (int)((local >> 4) & 31);
  size_t rest = local >> 9;
  int ksteps = K >> 5;
  int ks    = (int)(rest % ksteps);
  int ntile = (int)(rest / ksteps);
  int col = ntile * 16 + (lane & 15);
  int kk  = ks * 32 + (lane >> 4) * 8 + ((e < 8) ? e : e + 8);
  dst[local] = f2bf(src[(size_t)kk * N + col]);
}

__global__ void pack_module_kernel(const float* qw, const float* kw,
                                   const float* vw, const float* aow,
                                   const float* iw, const float* ow,
                                   void* ws, int m) {
  size_t idx = (size_t)blockIdx.x * blockDim.x + threadIdx.x;
  if (idx >= WSLAB) return;
  unsigned short* dst = Ureg(ws) + OFF_WPACK + (size_t)m * WSLAB;
  if      (idx < WK)  pack_one(qw,  dst + WQ,  idx - WQ,  HID, HID);
  else if (idx < WV)  pack_one(kw,  dst + WK,  idx - WK,  HID, HID);
  else if (idx < WAO) pack_one(vw,  dst + WV,  idx - WV,  HID, HID);
  else if (idx < WI)  pack_one(aow, dst + WAO, idx - WAO, HID, HID);
  else if (idx < WO)  pack_one(iw,  dst + WI,  idx - WI,  HID, FF);
  else                pack_one(ow,  dst + WO,  idx - WO,  FF,  HID);
}

__global__ void pack_single_kernel(const float* src, unsigned short* dst,
                                   int K, int N) {
  size_t idx = (size_t)blockIdx.x * blockDim.x + threadIdx.x;
  if (idx >= (size_t)K * N) return;
  pack_one(src, dst, idx, K, N);
}

__global__ void copy_params_kernel(const float* qb, const float* kb,
                                   const float* vb, const float* aob,
                                   const float* ib, const float* ob,
                                   const float* l1g, const float* l1b,
                                   const float* l2g, const float* l2b,
                                   float* dst) {
  int i = blockIdx.x * blockDim.x + threadIdx.x;
  if (i >= (int)PSLABF) return;
  const float* s; int off;
  if      (i < 768)  { s = qb;  off = i; }
  else if (i < 1536) { s = kb;  off = i - 768; }
  else if (i < 2304) { s = vb;  off = i - 1536; }
  else if (i < 3072) { s = aob; off = i - 2304; }
  else if (i < 6144) { s = ib;  off = i - 3072; }
  else if (i < 6912) { s = ob;  off = i - 6144; }
  else if (i < 7680) { s = l1g; off = i - 6912; }
  else if (i < 8448) { s = l1b; off = i - 7680; }
  else if (i < 9216) { s = l2g; off = i - 8448; }
  else               { s = l2b; off = i - 9216; }
  dst[i] = s[off];
}

__global__ void cvt_bf16_kernel(const float* src, unsigned short* dst, int n) {
  int i = blockIdx.x * blockDim.x + threadIdx.x;
  if (i < n) dst[i] = f2bf(src[i]);
}

// ----------------------------- GEMM kernels --------------------------------
__global__ void gemm_simple_kernel(const unsigned short* A,
                                   const unsigned short* Bp, const float* bias,
                                   float* Cf, unsigned short* Cbf,
                                   int K, int N, int ldc, int act) {
  gemm_wave_body(A, Bp, bias, Cf, Cbf, K, N, ldc, 0, act);
}

// which: 0=Q 1=K 2=V 3=AO 4=FFN-in(gelu->bf16) 5=FFN-out
__global__ void mod_gemm_kernel(void* ws, const int* args, int step, int which) {
  int b = blockIdx.z;
  const int* pa = args + (b * NSTEP + step) * 6;
  int fid = pa[0];
  unsigned short* U = Ureg(ws);
  float* F = Freg(ws);
  const unsigned short* wslab = U + OFF_WPACK + (size_t)fid * WSLAB;
  const float* pbk = F + OFF_PBLK + (size_t)fid * PSLABF;

  unsigned short* xbf   = U + OFF_XBF   + (size_t)b * 80 * HID;
  unsigned short* ctxbf = U + OFF_CTXBF + (size_t)b * 48 * HID;
  unsigned short* h1bf  = U + OFF_H1BF  + (size_t)b * 48 * HID;
  unsigned short* ffbf  = U + OFF_FFBF  + (size_t)b * 48 * FF;
  float* qkv  = F + OFF_QKV  + (size_t)b * 80 * (3 * HID);
  float* aout = F + OFF_AOUT + (size_t)b * 48 * HID;
  float* oout = F + OFF_OOUT + (size_t)b * 48 * HID;

  const unsigned short *A, *Bp; const float* bias;
  float* Cf = nullptr; unsigned short* Cbf = nullptr;
  int K, N, ldc, coloff = 0, act = 0;
  switch (which) {
    case 0: A = xbf;   Bp = wslab + WQ;  bias = pbk + PB_QB;  Cf = qkv;
            K = HID; N = HID; ldc = 3 * HID; coloff = 0;       break;
    case 1: A = xbf;   Bp = wslab + WK;  bias = pbk + PB_KB;  Cf = qkv;
            K = HID; N = HID; ldc = 3 * HID; coloff = HID;     break;
    case 2: A = xbf;   Bp = wslab + WV;  bias = pbk + PB_VB;  Cf = qkv;
            K = HID; N = HID; ldc = 3 * HID; coloff = 2 * HID; break;
    case 3: A = ctxbf; Bp = wslab + WAO; bias = pbk + PB_AOB; Cf = aout;
            K = HID; N = HID; ldc = HID;                       break;
    case 4: A = h1bf;  Bp = wslab + WI;  bias = pbk + PB_IB;  Cbf = ffbf;
            K = HID; N = FF;  ldc = FF;  act = 1;              break;
    default:A = ffbf;  Bp = wslab + WO;  bias = pbk + PB_OB;  Cf = oout;
            K = FF;  N = HID; ldc = HID;                       break;
  }
  gemm_wave_body(A, Bp, bias, Cf, Cbf, K, N, ldc, coloff, act);
}

// --------------------------- attention (fp32) ------------------------------
__global__ void attn_kernel(void* ws, const int* args, int step) {
  int h = blockIdx.x, b = blockIdx.y;
  const int* pa = args + (b * NSTEP + step) * 6;
  int S = (pa[2] > 0) ? 75 : 39;
  float* F = Freg(ws);
  const float* qkv = F + OFF_QKV + (size_t)b * 80 * (3 * HID);
  unsigned short* ctxbf = Ureg(ws) + OFF_CTXBF + (size_t)b * 48 * HID;

  __shared__ float ksh[80 * HD];
  __shared__ float vsh[80 * HD];
  __shared__ float qsh[LTOK * HD];
  __shared__ float scs[LTOK * 80];
  int tid = threadIdx.x;
  for (int idx = tid; idx < S * HD; idx += 64) {
    int r = idx >> 6, d = idx & 63;
    ksh[idx] = qkv[(size_t)r * (3 * HID) + HID     + h * HD + d];
    vsh[idx] = qkv[(size_t)r * (3 * HID) + 2 * HID + h * HD + d];
  }
  for (int idx = tid; idx < LTOK * HD; idx += 64) {
    int r = idx >> 6, d = idx & 63;
    qsh[idx] = qkv[(size_t)r * (3 * HID) + h * HD + d];
  }
  __syncthreads();
  if (tid < LTOK) {
    const float* q = qsh + tid * HD;
    float mx = -1e30f;
    for (int j = 0; j < S; ++j) {
      float s = 0.f;
      for (int d = 0; d < HD; ++d) s += q[d] * ksh[j * HD + d];
      s *= 0.125f;                         // 1/sqrt(64)
      scs[tid * 80 + j] = s;
      mx = fmaxf(mx, s);
    }
    float sum = 0.f;
    for (int j = 0; j < S; ++j) {
      float e = expf(scs[tid * 80 + j] - mx);
      scs[tid * 80 + j] = e;
      sum += e;
    }
    float inv = 1.f / sum;
    for (int d = 0; d < HD; ++d) {
      float c = 0.f;
      for (int j = 0; j < S; ++j) c += scs[tid * 80 + j] * vsh[j * HD + d];
      ctxbf[(size_t)tid * HID + h * HD + d] = f2bf(c * inv);
    }
  }
}

// -------------------------------- layernorm --------------------------------
__global__ void ln_kernel(void* ws, const int* args, int step, int which) {
  int row = blockIdx.x, b = blockIdx.y;
  const int* pa = args + (b * NSTEP + step) * 6;
  int fid = pa[0];
  float* F = Freg(ws);
  unsigned short* U = Ureg(ws);
  const float* pbk = F + OFF_PBLK + (size_t)fid * PSLABF;

  const float *g, *bt, *in1, *res;
  float* outf; unsigned short* outbf = nullptr;
  if (which == 1) {
    g = pbk + PB_L1G; bt = pbk + PB_L1B;
    in1 = F + OFF_AOUT + (size_t)b * 48 * HID;
    res = F + OFF_XF   + (size_t)b * 80 * HID;
    outf  = F + OFF_H1F  + (size_t)b * 48 * HID;
    outbf = U + OFF_H1BF + (size_t)b * 48 * HID;
  } else {
    g = pbk + PB_L2G; bt = pbk + PB_L2B;
    in1 = F + OFF_OOUT + (size_t)b * 48 * HID;
    res = F + OFF_H1F  + (size_t)b * 48 * HID;
    outf = F + OFF_HS + ((size_t)b * 26 + (step + 1)) * LTOK * HID;
  }
  __shared__ float buf[HID];
  __shared__ float r1[256];
  __shared__ float r2[256];
  int tid = threadIdx.x;
  float s1 = 0.f, s2 = 0.f;
  for (int c = tid; c < HID; c += 256) {
    float t = in1[(size_t)row * HID + c] + res[(size_t)row * HID + c];
    buf[c] = t; s1 += t; s2 += t * t;
  }
  r1[tid] = s1; r2[tid] = s2;
  __syncthreads();
  for (int off = 128; off > 0; off >>= 1) {
    if (tid < off) { r1[tid] += r1[tid + off]; r2[tid] += r2[tid + off]; }
    __syncthreads();
  }
  float mean = r1[0] * (1.f / HID);
  float var  = r2[0] * (1.f / HID) - mean * mean;
  float inv  = rsqrtf(var + 1e-12f);
  for (int c = tid; c < HID; c += 256) {
    float v = (buf[c] - mean) * inv * g[c] + bt[c];
    outf[(size_t)row * HID + c] = v;
    if (outbf) outbf[(size_t)row * HID + c] = f2bf(v);
  }
}

// ------------------------- module input assembly ---------------------------
__global__ void build_x_kernel(void* ws, const int* args, int step,
                               const float* arg_emb, const float* pos_emb,
                               const float* tt_emb) {
  int row = blockIdx.x, b = blockIdx.y;
  const int* pa = args + (b * NSTEP + step) * 6;
  int i0 = pa[1], i1 = pa[2];
  bool two = i1 > 0;
  float* F = Freg(ws);
  const float* hs = F + OFF_HS;
  float* xf = F + OFF_XF + (size_t)b * 80 * HID;
  unsigned short* xbf = Ureg(ws) + OFF_XBF + (size_t)b * 80 * HID;
  int tid = threadIdx.x;
  for (int c = tid; c < HID; c += 256) {
    float v = 0.f;
    if (two) {
      if (row < 36)
        v = hs[(((size_t)b * 26 + i0) * LTOK + row) * HID + c] + tt_emb[c];
      else if (row < 72)
        v = hs[(((size_t)b * 26 + i1) * LTOK + (row - 36)) * HID + c] +
            tt_emb[HID + c];
      else if (row < 75) {
        int j = row - 72;
        v = arg_emb[(size_t)pa[3 + j] * HID + c] +
            pos_emb[(size_t)(LTOK + j) * HID + c];
      }
    } else {
      if (row < 36)
        v = hs[(((size_t)b * 26 + i0) * LTOK + row) * HID + c];
      else if (row < 39) {
        int j = row - 36;
        v = arg_emb[(size_t)pa[3 + j] * HID + c] +
            pos_emb[(size_t)(LTOK + j) * HID + c];
      }
    }
    xf[(size_t)row * HID + c] = v;
    xbf[(size_t)row * HID + c] = f2bf(v);
  }
}

// ------------------------ embedding finish (loc + LN + pos) ----------------
__global__ void embed_finish_kernel(void* ws, const float* spatials,
                                    const float* locw, const float* locb,
                                    const float* g, const float* bt,
                                    const float* pos_emb) {
  int row = blockIdx.x;           // 0..287  (b*36 + l)
  int b = row / LTOK, l = row % LTOK;
  float* F = Freg(ws);
  const float* et = F + OFF_EMBT + (size_t)row * HID;
  float* out = F + OFF_HS + (((size_t)b * 26 + 0) * LTOK + l) * HID;
  __shared__ float buf[HID];
  __shared__ float r1[256];
  __shared__ float r2[256];
  float sp[5];
#pragma unroll
  for (int k2 = 0; k2 < 5; ++k2) sp[k2] = spatials[(size_t)row * 5 + k2];
  int tid = threadIdx.x;
  float s1 = 0.f, s2 = 0.f;
  for (int c = tid; c < HID; c += 256) {
    float t = et[c] + locb[c];
#pragma unroll
    for (int k2 = 0; k2 < 5; ++k2) t += sp[k2] * locw[(size_t)k2 * HID + c];
    buf[c] = t; s1 += t; s2 += t * t;
  }
  r1[tid] = s1; r2[tid] = s2;
  __syncthreads();
  for (int off = 128; off > 0; off >>= 1) {
    if (tid < off) { r1[tid] += r1[tid + off]; r2[tid] += r2[tid + off]; }
    __syncthreads();
  }
  float mean = r1[0] * (1.f / HID);
  float var  = r2[0] * (1.f / HID) - mean * mean;
  float inv  = rsqrtf(var + 1e-12f);
  for (int c = tid; c < HID; c += 256)
    out[c] = (buf[c] - mean) * inv * g[c] + bt[c] +
             pos_emb[(size_t)l * HID + c];
}

// ------------------------------ classifier ---------------------------------
__global__ void gather_cls_kernel(void* ws) {
  int row = blockIdx.x;  // 0..15
  float* F = Freg(ws);
  unsigned short* dst = Ureg(ws) + OFF_CLSINBF + (size_t)row * HID;
  int tid = threadIdx.x;
  for (int c = tid; c < HID; c += 256) {
    float v = (row < BS)
                  ? F[OFF_HS + (((size_t)row * 26 + 25) * LTOK + 0) * HID + c]
                  : 0.f;
    dst[c] = f2bf(v);
  }
}

__global__ void write_out_kernel(void* ws, float* out) {
  size_t i = (size_t)blockIdx.x * blockDim.x + threadIdx.x;
  float* F = Freg(ws);
  const size_t NOUT = (size_t)BS * LTOK * HID;  // 221184
  if (i < NOUT) {
    size_t b = i / (LTOK * HID), rem = i % (LTOK * HID);
    out[i] = F[OFF_HS + ((b * 26 + 25) * (size_t)(LTOK * HID)) + rem];
  } else if (i < NOUT + (size_t)BS * 32) {
    size_t j = i - NOUT;
    out[i] = F[OFF_PRED + (j / 32) * 32 + (j % 32)];
  }
}

// ------------------------------- host side ---------------------------------
extern "C" void kernel_launch(void* const* d_in, const int* in_sizes, int n_in,
                              void* d_out, int out_size, void* d_ws,
                              size_t ws_size, hipStream_t stream) {
  (void)in_sizes; (void)n_in; (void)out_size; (void)ws_size;
  const float* features = (const float*)d_in[0];
  const float* spatials = (const float*)d_in[1];
  // d_in[2] = image_mask (unused by the reference math)
  const int*   args     = (const int*)d_in[3];
  const float* imgw  = (const float*)d_in[4];
  const float* imgb  = (const float*)d_in[5];
  const float* locw  = (const float*)d_in[6];
  const float* locb  = (const float*)d_in[7];
  const float* ilng  = (const float*)d_in[8];
  const float* ilnb  = (const float*)d_in[9];
  const float* argem = (const float*)d_in[10];
  const float* posem = (const float*)d_in[11];
  const float* ttem  = (const float*)d_in[12];
  const float* cls1w = (const float*)d_in[13];
  const float* cls1b = (const float*)d_in[14];
  const float* cls2w = (const float*)d_in[15];
  const float* cls2b = (const float*)d_in[16];
  auto modp = [&](int m, int j) { return (const float*)d_in[17 + m * 16 + j]; };

  unsigned short* U = (unsigned short*)d_ws;
  float* Fh = (float*)((char*)d_ws + FBYTE_BASE);

  // --- pack all weights to bf16 WMMA fragment layout (per launch) ---
  for (int m = 0; m < NMOD; ++m) {
    pack_module_kernel<<<(unsigned)((WSLAB + 255) / 256), 256, 0, stream>>>(
        modp(m, 0), modp(m, 2), modp(m, 4), modp(m, 6), modp(m, 10),
        modp(m, 12), d_ws, m);
    copy_params_kernel<<<(unsigned)((PSLABF + 255) / 256), 256, 0, stream>>>(
        modp(m, 1), modp(m, 3), modp(m, 5), modp(m, 7), modp(m, 11),
        modp(m, 13), modp(m, 8), modp(m, 9), modp(m, 14), modp(m, 15),
        Fh + OFF_PBLK + (size_t)m * PSLABF);
  }
  pack_single_kernel<<<(VFEAT * HID + 255) / 256, 256, 0, stream>>>(
      imgw, U + OFF_IMGWP, VFEAT, HID);
  pack_single_kernel<<<(HID * 2 * HID + 255) / 256, 256, 0, stream>>>(
      cls1w, U + OFF_CLS1P, HID, 2 * HID);
  pack_single_kernel<<<(2 * HID * 32 + 255) / 256, 256, 0, stream>>>(
      cls2w, U + OFF_CLS2P, 2 * HID, 32);

  // --- embedding: hidden = LN(feat@Wimg + b + spat@Wloc + b) + pos ---
  int nfeat = BS * LTOK * VFEAT;
  cvt_bf16_kernel<<<(nfeat + 255) / 256, 256, 0, stream>>>(
      features, U + OFF_FEATBF, nfeat);
  gemm_simple_kernel<<<dim3(3, 18, 1), 256, 0, stream>>>(
      U + OFF_FEATBF, U + OFF_IMGWP, imgb, Fh + OFF_EMBT, nullptr,
      VFEAT, HID, HID, 0);
  embed_finish_kernel<<<BS * LTOK, 256, 0, stream>>>(
      d_ws, spatials, locw, locb, ilng, ilnb, posem);

  // --- 25 program steps, all 8 batch lanes in parallel (grid.z = b) ---
  for (int step = 0; step < NSTEP; ++step) {
    build_x_kernel<<<dim3(80, BS), 256, 0, stream>>>(d_ws, args, step, argem,
                                                     posem, ttem);
    mod_gemm_kernel<<<dim3(3, 3, BS), 256, 0, stream>>>(d_ws, args, step, 0);
    mod_gemm_kernel<<<dim3(3, 5, BS), 256, 0, stream>>>(d_ws, args, step, 1);
    mod_gemm_kernel<<<dim3(3, 5, BS), 256, 0, stream>>>(d_ws, args, step, 2);
    attn_kernel<<<dim3(HEADS, BS), 64, 0, stream>>>(d_ws, args, step);
    mod_gemm_kernel<<<dim3(3, 3, BS), 256, 0, stream>>>(d_ws, args, step, 3);
    ln_kernel<<<dim3(LTOK, BS), 256, 0, stream>>>(d_ws, args, step, 1);
    mod_gemm_kernel<<<dim3(12, 3, BS), 256, 0, stream>>>(d_ws, args, step, 4);
    mod_gemm_kernel<<<dim3(3, 3, BS), 256, 0, stream>>>(d_ws, args, step, 5);
    ln_kernel<<<dim3(LTOK, BS), 256, 0, stream>>>(d_ws, args, step, 2);
  }

  // --- classifier head ---
  gather_cls_kernel<<<16, 256, 0, stream>>>(d_ws);
  gemm_simple_kernel<<<dim3(6, 1), 256, 0, stream>>>(
      U + OFF_CLSINBF, U + OFF_CLS1P, cls1b, nullptr, U + OFF_CLSMIDBF,
      HID, 2 * HID, 2 * HID, 2);
  gemm_simple_kernel<<<dim3(1, 1), 256, 0, stream>>>(
      U + OFF_CLSMIDBF, U + OFF_CLS2P, cls2b, Fh + OFF_PRED, nullptr,
      2 * HID, 32, 32, 0);

  int ntot = BS * LTOK * HID + BS * 32;
  write_out_kernel<<<(ntot + 255) / 256, 256, 0, stream>>>(d_ws,
                                                           (float*)d_out);
}
